// GPS_16149077033391
// MI455X (gfx1250) — matrix operand order repaired and checked by reading, not compiled, not run on hardware
//
#include <hip/hip_runtime.h>
#include <hip/hip_bf16.h>

// ---------------------------------------------------------------------------
// Problem constants (match reference)
// ---------------------------------------------------------------------------
#define Nn   65536           // nodes
#define Cc   128             // channels
#define PEd  16
#define Ll   4
#define Hh   4
#define DHh  32              // Cc / Hh
#define Gg   256             // graphs
#define Ss   256             // nodes per graph
#define Ee   524288          // edges
#define C2   256             // 2*C

typedef __bf16 bf16_t;
typedef __attribute__((ext_vector_type(16))) __bf16 v16bf;
typedef __attribute__((ext_vector_type(8)))  float  v8f;
typedef unsigned int u32x4 __attribute__((ext_vector_type(4)));
typedef int          i32x8 __attribute__((ext_vector_type(8)));
typedef int          i32x4 __attribute__((ext_vector_type(4)));

// ---------------------------------------------------------------------------
// Wave helpers (wave32; reductions within 16-lane row groups)
// ---------------------------------------------------------------------------
__device__ __forceinline__ float rowmax16(float v) {
    #pragma unroll
    for (int m = 1; m < 16; m <<= 1) v = fmaxf(v, __shfl_xor(v, m, 16));
    return v;
}
__device__ __forceinline__ float rowsum16(float v) {
    #pragma unroll
    for (int m = 1; m < 16; m <<= 1) v += __shfl_xor(v, m, 16);
    return v;
}

// ---------------------------------------------------------------------------
// Weight pre-transpose: W[K][Nc] f32 -> Wt[Nc][K] bf16 (done once per launch)
// Makes the TDM tile rows equal to B-fragment K-runs (contiguous LDS reads).
// ---------------------------------------------------------------------------
__global__ void convert_wT_kernel(const float* __restrict__ W, bf16_t* __restrict__ Wt,
                                  int K, int Nc)
{
    int id = blockIdx.x * 256 + threadIdx.x;
    if (id >= K * Nc) return;
    int k = id / Nc, n = id - k * Nc;
    Wt[(size_t)n * K + k] = (bf16_t)W[id];
}

// ---------------------------------------------------------------------------
// WMMA GEMM:  Y[M,Nc] = act( X[M,K] @ W[K,Nc] + bias (+ R) )
// Weights come pre-transposed bf16 [Nc][K].  Block = 8 waves, 256x64 output
// macro-tile; each wave owns 32(M)x64(N) = 2x4 accumulators -> 8 WMMA / K-step.
// W tile (64 x K bf16) is DMA'd into LDS by the Tensor Data Mover with
// per-row padding of 4 DWORDs (KP = K+8 halves) for conflict-free B reads.
// ---------------------------------------------------------------------------
__global__ __launch_bounds__(256) void wmma_gemm_kernel(
    const float* __restrict__ X, const bf16_t* __restrict__ Wt,
    const float* __restrict__ bias, const float* __restrict__ R,
    float* __restrict__ Y, int M, int K, int Nc, int relu)
{
    __shared__ bf16_t sW[64 * (256 + 8)];         // [col][K+8] bf16, max K = 256
    const int KP  = K + 8;
    const int n0  = blockIdx.x * 64;
    const int m0  = blockIdx.y * 256;
    const int tid = threadIdx.x;

    // ---- TDM: stage Wt[n0:n0+64][0:K] into LDS (wave 0 issues the DMA) ----
    if (tid < 32) {
        const unsigned long long ga =
            (unsigned long long)(const void*)(Wt + (size_t)n0 * K);
        const unsigned int lds = (unsigned int)(unsigned long long)(void*)sW;
        const int rowDw      = K >> 1;            // dwords per tile row
        const int padIntCode = (rowDw == 64) ? 5 : 6;   // 64 or 128 dwords
        u32x4 g0;
        g0[0] = 1u;                                              // count = 1
        g0[1] = lds;                                             // lds_addr
        g0[2] = (unsigned int)(ga & 0xffffffffull);              // global_addr lo
        g0[3] = (unsigned int)((ga >> 32) & 0x1ffffffull)        // global_addr hi
              | (2u << 30);                                      // type = 2 (image)
        i32x8 g1;
        g1[0] = (1 << 16)                                        // data_size = 2B
              | (1 << 20)                                        // pad_enable
              | (padIntCode << 22)                               // pad_interval
              | (3 << 25);                                       // pad_amount = 4 dw
        g1[1] = (K & 0xffff) << 16;                              // tensor_dim0 lo16
        g1[2] = (K >> 16) | ((Nc & 0xffff) << 16);               // dim0 hi | dim1 lo
        g1[3] = (Nc >> 16) | (K << 16);                          // dim1 hi | tile_dim0=K
        g1[4] = 64;                                              // tile_dim1 = 64
        g1[5] = K;                                               // dim0_stride lo32
        g1[6] = 0;
        g1[7] = 0;
        i32x4 gz  = {0, 0, 0, 0};
        i32x8 gz8 = {0, 0, 0, 0, 0, 0, 0, 0};
        __builtin_amdgcn_tensor_load_to_lds(g0, g1, gz, gz, gz8, 0);
        __builtin_amdgcn_s_wait_tensorcnt(0);
    }
    __syncthreads();

    const int wave  = tid >> 5;
    const int lane  = tid & 31;
    const int lhalf = lane >> 4;
    const int l16   = lane & 15;
    const int rbase = m0 + wave * 32;

    v8f acc[2][4];
    #pragma unroll
    for (int mi = 0; mi < 2; ++mi)
        #pragma unroll
        for (int t = 0; t < 4; ++t) acc[mi][t] = (v8f){};

    for (int k0 = 0; k0 < K; k0 += 32) {
        v16bf a[2];
        #pragma unroll
        for (int mi = 0; mi < 2; ++mi) {
            const float* xp = X + (size_t)(rbase + mi * 16 + l16) * K + k0 + lhalf * 8;
            #pragma unroll
            for (int i = 0; i < 8; ++i) a[mi][i]     = (bf16_t)xp[i];       // K=k0+8lh+i
            #pragma unroll
            for (int i = 0; i < 8; ++i) a[mi][8 + i] = (bf16_t)xp[16 + i];  // +16
        }
        #pragma unroll
        for (int t = 0; t < 4; ++t) {
            const bf16_t* bp = sW + (t * 16 + l16) * KP + k0 + lhalf * 8;
            v16bf b;
            #pragma unroll
            for (int i = 0; i < 16; ++i) b[i] = bp[i < 8 ? i : i + 8];      // K=k0+8lh+i, +16
            acc[0][t] = __builtin_amdgcn_wmma_f32_16x16x32_bf16(false, a[0], false, b,
                                                                (short)0, acc[0][t],
                                                                false, false);
            acc[1][t] = __builtin_amdgcn_wmma_f32_16x16x32_bf16(false, a[1], false, b,
                                                                (short)0, acc[1][t],
                                                                false, false);
        }
    }

    #pragma unroll
    for (int t = 0; t < 4; ++t) {
        const int col = n0 + t * 16 + l16;
        const float bv = bias ? bias[col] : 0.f;
        #pragma unroll
        for (int mi = 0; mi < 2; ++mi) {
            #pragma unroll
            for (int r = 0; r < 8; ++r) {
                int rr = rbase + mi * 16 + lhalf * 8 + r;   // C-layout M = r + 8*lhalf
                float v = acc[mi][t][r] + bv;
                if (R)    v += R[(size_t)rr * Nc + col];
                if (relu) v  = fmaxf(v, 0.f);
                Y[(size_t)rr * Nc + col] = v;
            }
        }
    }
}

// ---------------------------------------------------------------------------
// Flash attention: one wave per (graph, head, 16-row q tile).
// q,k,v,o are [N, C] = [G*S, H*DH] f32.  scores K-dim = DH = 32 (one WMMA).
// ---------------------------------------------------------------------------
__global__ __launch_bounds__(128) void attn_kernel(
    const float* __restrict__ q, const float* __restrict__ k,
    const float* __restrict__ v, float* __restrict__ o)
{
    __shared__ bf16_t sP[4][16 * 16];                    // per-wave P bounce tile
    const int tid   = threadIdx.x;
    const int wave  = tid >> 5;
    const int lane  = tid & 31;
    const int lhalf = lane >> 4;
    const int l16   = lane & 15;
    const int wid   = blockIdx.x * 4 + wave;             // 0 .. G*H*(S/16)-1
    const int g     = wid >> 6;                          // H*(S/16) = 64
    const int hh    = (wid >> 4) & (Hh - 1);
    const int qt    = wid & 15;
    const float scale = 0.17677669529663687f;            // 1/sqrt(32)

    v16bf qa;
    {
        const float* qp = q + ((size_t)(g * Ss + qt * 16 + l16) * Hh + hh) * DHh + lhalf * 8;
        #pragma unroll
        for (int i = 0; i < 8; ++i) qa[i]     = (bf16_t)qp[i];
        #pragma unroll
        for (int i = 0; i < 8; ++i) qa[8 + i] = (bf16_t)qp[16 + i];
    }

    float m[8], l[8];
    v8f acc0 = {}, acc1 = {};
    #pragma unroll
    for (int r = 0; r < 8; ++r) { m[r] = -1e30f; l[r] = 0.f; }

    for (int kt = 0; kt < Ss / 16; ++kt) {
        v16bf kb;
        {
            const float* kp = k + ((size_t)(g * Ss + kt * 16 + l16) * Hh + hh) * DHh + lhalf * 8;
            #pragma unroll
            for (int i = 0; i < 16; ++i) kb[i] = (bf16_t)kp[i < 8 ? i : i + 8];
        }
        v8f s = {};
        s = __builtin_amdgcn_wmma_f32_16x16x32_bf16(false, qa, false, kb,
                                                    (short)0, s, false, false);
        #pragma unroll
        for (int r = 0; r < 8; ++r) {
            float sv = s[r] * scale;
            float mn = fmaxf(m[r], rowmax16(sv));
            float al = __expf(m[r] - mn);
            float p  = __expf(sv - mn);
            l[r] = l[r] * al + rowsum16(p);
            acc0[r] *= al; acc1[r] *= al;
            m[r] = mn;
            sP[wave][(lhalf * 8 + r) * 16 + l16] = (bf16_t)p;   // [M][key]
        }
        v16bf pa;                                   // P as A-frag, K 16..31 zero
        #pragma unroll
        for (int i = 0; i < 8; ++i) pa[i]     = sP[wave][l16 * 16 + lhalf * 8 + i];
        #pragma unroll
        for (int i = 0; i < 8; ++i) pa[8 + i] = (bf16_t)0.f;
        v16bf vb0, vb1;                             // V frags, K rows 16..31 zero
        {
            const float* vp = v + ((size_t)(g * Ss + kt * 16 + lhalf * 8) * Hh + hh) * DHh;
            #pragma unroll
            for (int i = 0; i < 8; ++i) {
                vb0[i] = (bf16_t)vp[(size_t)i * Cc + l16];
                vb1[i] = (bf16_t)vp[(size_t)i * Cc + 16 + l16];
            }
            #pragma unroll
            for (int i = 8; i < 16; ++i) { vb0[i] = (bf16_t)0.f; vb1[i] = (bf16_t)0.f; }
        }
        acc0 = __builtin_amdgcn_wmma_f32_16x16x32_bf16(false, pa, false, vb0,
                                                       (short)0, acc0, false, false);
        acc1 = __builtin_amdgcn_wmma_f32_16x16x32_bf16(false, pa, false, vb1,
                                                       (short)0, acc1, false, false);
    }
    #pragma unroll
    for (int r = 0; r < 8; ++r) {
        float inv = 1.f / l[r];
        int node  = g * Ss + qt * 16 + lhalf * 8 + r;
        float* op = o + (size_t)node * Cc + hh * DHh;
        op[l16]      = acc0[r] * inv;
        op[16 + l16] = acc1[r] * inv;
    }
}

// ---------------------------------------------------------------------------
// BatchNorm (training-mode batch stats): two-pass with atomics
// ---------------------------------------------------------------------------
__global__ void bn_stats_kernel(const float* __restrict__ x, float* __restrict__ stats,
                                int rows, int cdim, int rpb)
{
    int c = threadIdx.x;
    if (c >= cdim) return;
    int r0 = blockIdx.x * rpb;
    int r1 = min(r0 + rpb, rows);
    float s = 0.f, s2 = 0.f;
    for (int r = r0; r < r1; ++r) {
        float vv = x[(size_t)r * cdim + c];
        s += vv; s2 += vv * vv;
    }
    atomicAdd(&stats[c], s);
    atomicAdd(&stats[cdim + c], s2);
}

__global__ void bn_apply_kernel(const float* __restrict__ x, float* __restrict__ y,
                                const float* __restrict__ stats,
                                const float* __restrict__ g, const float* __restrict__ b,
                                float invRows, int cdim, int relu, int total)
{
    int id = blockIdx.x * 256 + threadIdx.x;
    if (id >= total) return;
    int c = id % cdim;
    float mu  = stats[c] * invRows;
    float var = stats[cdim + c] * invRows - mu * mu;
    float vv  = (x[id] - mu) * rsqrtf(var + 1e-5f) * g[c] + b[c];
    if (relu) vv = fmaxf(vv, 0.f);
    y[id] = vv;
}

// ---------------------------------------------------------------------------
// Input embeddings
// ---------------------------------------------------------------------------
__global__ void embed_node_kernel(const float* __restrict__ x, const float* __restrict__ pe,
                                  const float* __restrict__ stats,
                                  const float* __restrict__ png, const float* __restrict__ pnb,
                                  const float* __restrict__ nw, const float* __restrict__ nb,
                                  const float* __restrict__ pw, const float* __restrict__ pb,
                                  float* __restrict__ h, float invRows)
{
    int id = blockIdx.x * 256 + threadIdx.x;
    if (id >= Nn * Cc) return;
    int node = id / Cc, c = id % Cc;
    float acc;
    if (c < Cc - PEd) {
        acc = nb[c];
        #pragma unroll
        for (int j = 0; j < 10; ++j)
            acc += x[node * 10 + j] * nw[j * (Cc - PEd) + c];
    } else {
        int cc = c - (Cc - PEd);
        acc = pb[cc];
        #pragma unroll
        for (int j = 0; j < 10; ++j) {
            float mu  = stats[j] * invRows;
            float var = stats[10 + j] * invRows - mu * mu;
            float xn  = (pe[node * 10 + j] - mu) * rsqrtf(var + 1e-5f) * png[j] + pnb[j];
            acc += xn * pw[j * PEd + cc];
        }
    }
    h[id] = acc;
}

__global__ void embed_edge_kernel(const float* __restrict__ ea, const float* __restrict__ ew,
                                  const float* __restrict__ eb, float* __restrict__ e)
{
    int id = blockIdx.x * 256 + threadIdx.x;            // over E*C
    int eid = id / Cc, c = id % Cc;
    float acc = eb[c];
    #pragma unroll
    for (int j = 0; j < 16; ++j)
        acc += ea[eid * 16 + j] * ew[j * Cc + c];
    e[id] = acc;
}

// ---------------------------------------------------------------------------
// GINE: z += relu(h[src] + e) scattered to dst (z pre-initialized with h)
// ---------------------------------------------------------------------------
__global__ void gine_msg_kernel(const float* __restrict__ h, const float* __restrict__ e,
                                const int* __restrict__ src, const int* __restrict__ dst,
                                float* __restrict__ z)
{
    int eid = blockIdx.x * 2 + (threadIdx.x >> 7);
    int c   = threadIdx.x & (Cc - 1);
    int s = src[eid], d = dst[eid];
    float m = h[(size_t)s * Cc + c] + e[(size_t)eid * Cc + c];
    m = fmaxf(m, 0.f);
    atomicAdd(&z[(size_t)d * Cc + c], m);
}

// ---------------------------------------------------------------------------
// Elementwise helpers
// ---------------------------------------------------------------------------
__global__ void copy_kernel(const float* __restrict__ a, float* __restrict__ y, int n)
{
    int id = blockIdx.x * 256 + threadIdx.x;
    if (id < n) y[id] = a[id];
}
__global__ void add_kernel(const float* __restrict__ a, const float* __restrict__ b,
                           float* __restrict__ y, int n)
{
    int id = blockIdx.x * 256 + threadIdx.x;
    if (id < n) y[id] = a[id] + b[id];
}
__global__ void assemble_h_kernel(const float* __restrict__ h, float* __restrict__ out)
{
    int id = blockIdx.x * 256 + threadIdx.x;
    if (id >= Nn * Cc) return;
    int node = id / Cc, c = id % Cc;
    out[(size_t)node * (Cc + 4) + c] = h[id];
}
__global__ void head_out_kernel(const float* __restrict__ a2, const float* __restrict__ w3,
                                const float* __restrict__ b3, float* __restrict__ out,
                                int od, int colbase)
{
    int node = blockIdx.x * 256 + threadIdx.x;
    if (node >= Nn) return;
    for (int o2 = 0; o2 < od; ++o2) {
        float acc = b3[o2];
        for (int j = 0; j < 128; ++j)
            acc += a2[(size_t)node * 128 + j] * w3[j * od + o2];
        out[(size_t)node * (Cc + 4) + colbase + o2] = acc;
    }
}

// ---------------------------------------------------------------------------
// Host orchestration
// ---------------------------------------------------------------------------
extern "C" void kernel_launch(void* const* d_in, const int* in_sizes, int n_in,
                              void* d_out, int out_size, void* d_ws, size_t ws_size,
                              hipStream_t stream)
{
    (void)in_sizes; (void)n_in; (void)out_size; (void)ws_size;

    auto P = [&](int i) { return (const float*)d_in[i]; };
    enum {
        I_PENG = 3, I_PENB, I_NW, I_NB, I_PW, I_PB, I_EW, I_EB,
        I_GW1, I_GW2, I_WQ, I_WK, I_WV, I_WO, I_FW1, I_FW2,
        I_GB1, I_GB2, I_BQ, I_BK, I_BV, I_BO, I_FB1, I_FB2,
        I_N1G, I_N1B, I_N2G, I_N2B, I_N3G, I_N3B,
        I_HEAD0 = 33,                 // p1_w1 ... (10 entries per head)
        I_EIDX = 63, I_BATCH = 64
    };
    const float* x_in  = P(0);
    const float* pe_in = P(1);
    const float* ea_in = P(2);
    const int* ei = (const int*)d_in[I_EIDX];
    float* out = (float*)d_out;

    // -- workspace bump allocator --
    char* wsb = (char*)d_ws;
    size_t off = 0;
    auto allocf = [&](size_t n) { float* p = (float*)(wsb + off); off += n * 4; return p; };
    auto allocb = [&](size_t n) { bf16_t* p = (bf16_t*)(wsb + off); off += ((n * 2 + 255) & ~255ull); return p; };
    float* e_buf = allocf((size_t)Ee * Cc);
    float* h     = allocf((size_t)Nn * Cc);
    float* z     = allocf((size_t)Nn * Cc);
    float* t1    = allocf((size_t)Nn * Cc);
    float* h1    = allocf((size_t)Nn * Cc);
    float* qb    = allocf((size_t)Nn * Cc);
    float* kb    = allocf((size_t)Nn * Cc);
    float* vb    = allocf((size_t)Nn * Cc);
    float* ab    = allocf((size_t)Nn * Cc);
    float* h2    = allocf((size_t)Nn * Cc);
    float* outc  = allocf((size_t)Nn * Cc);
    float* f1    = allocf((size_t)Nn * C2);
    float* a1    = allocf((size_t)Nn * 128);
    float* a2    = allocf((size_t)Nn * 128);
    float* stats = allocf(512);

    // bf16-transposed weights: per layer {gw1,gw2,wq,wk,wv,wo,fw1,fw2}, heads {w1,w2}
    bf16_t* WT[Ll][8];
    for (int l = 0; l < Ll; ++l) {
        for (int wsel = 0; wsel < 6; ++wsel) WT[l][wsel] = allocb((size_t)Cc * Cc);
        WT[l][6] = allocb((size_t)Cc * C2);
        WT[l][7] = allocb((size_t)C2 * Cc);
    }
    bf16_t* WTh[3][2];
    for (int hi = 0; hi < 3; ++hi)
        for (int wsel = 0; wsel < 2; ++wsel) WTh[hi][wsel] = allocb((size_t)128 * 128);

    auto cvtW = [&](const float* W, bf16_t* Wt, int K, int Nc) {
        convert_wT_kernel<<<(K * Nc + 255) / 256, 256, 0, stream>>>(W, Wt, K, Nc);
    };
    auto gemm = [&](const float* X, const bf16_t* Wt, const float* bias, const float* R,
                    float* Y, int M, int K, int Nc, int relu) {
        wmma_gemm_kernel<<<dim3(Nc / 64, M / 256), 256, 0, stream>>>(X, Wt, bias, R, Y,
                                                                     M, K, Nc, relu);
    };
    auto bn = [&](const float* xin, float* yout, const float* g, const float* b,
                  int rows, int cdim, int relu) {
        (void)hipMemsetAsync(stats, 0, 2 * cdim * sizeof(float), stream);
        bn_stats_kernel<<<(rows + 255) / 256, cdim, 0, stream>>>(xin, stats, rows, cdim, 256);
        int total = rows * cdim;
        bn_apply_kernel<<<(total + 255) / 256, 256, 0, stream>>>(xin, yout, stats, g, b,
                                                                 1.f / rows, cdim, relu, total);
    };

    // ---------------- one-time weight transposition to bf16 ----------------
    const int wIdx[8] = {I_GW1, I_GW2, I_WQ, I_WK, I_WV, I_WO, I_FW1, I_FW2};
    for (int l = 0; l < Ll; ++l) {
        for (int wsel = 0; wsel < 6; ++wsel)
            cvtW(P(wIdx[wsel]) + (size_t)l * Cc * Cc, WT[l][wsel], Cc, Cc);
        cvtW(P(I_FW1) + (size_t)l * Cc * C2, WT[l][6], Cc, C2);
        cvtW(P(I_FW2) + (size_t)l * C2 * Cc, WT[l][7], C2, Cc);
    }
    for (int hi = 0; hi < 3; ++hi) {
        int base = I_HEAD0 + hi * 10;
        cvtW(P(base + 0), WTh[hi][0], Cc, 128);
        cvtW(P(base + 4), WTh[hi][1], 128, 128);
    }

    // ---------------- embeddings ----------------
    (void)hipMemsetAsync(stats, 0, 20 * sizeof(float), stream);
    bn_stats_kernel<<<(Nn + 255) / 256, 10, 0, stream>>>(pe_in, stats, Nn, 10, 256);
    embed_node_kernel<<<(Nn * Cc) / 256, 256, 0, stream>>>(
        x_in, pe_in, stats, P(I_PENG), P(I_PENB), P(I_NW), P(I_NB), P(I_PW), P(I_PB),
        h, 1.f / Nn);
    embed_edge_kernel<<<(Ee * Cc) / 256, 256, 0, stream>>>(ea_in, P(I_EW), P(I_EB), e_buf);

    // ---------------- layers ----------------
    for (int l = 0; l < Ll; ++l) {
        const float* gb1 = P(I_GB1) + l * Cc;
        const float* gb2 = P(I_GB2) + l * Cc;
        const float* bq  = P(I_BQ)  + l * Cc;
        const float* bk  = P(I_BK)  + l * Cc;
        const float* bv  = P(I_BV)  + l * Cc;
        const float* bo  = P(I_BO)  + l * Cc;
        const float* fb1 = P(I_FB1) + l * C2;
        const float* fb2 = P(I_FB2) + l * Cc;

        // GINE branch: z = h + sum relu(h[src] + e)
        copy_kernel<<<(Nn * Cc) / 256, 256, 0, stream>>>(h, z, Nn * Cc);
        gine_msg_kernel<<<Ee / 2, 256, 0, stream>>>(h, e_buf, ei, ei + Ee, z);
        gemm(z,  WT[l][0], gb1, nullptr, t1, Nn, Cc, Cc, 1);
        gemm(t1, WT[l][1], gb2, h,       h1, Nn, Cc, Cc, 0);     // + residual h
        bn(h1, h1, P(I_N1G) + l * Cc, P(I_N1B) + l * Cc, Nn, Cc, 0);

        // attention branch on h
        gemm(h, WT[l][2], bq, nullptr, qb, Nn, Cc, Cc, 0);
        gemm(h, WT[l][3], bk, nullptr, kb, Nn, Cc, Cc, 0);
        gemm(h, WT[l][4], bv, nullptr, vb, Nn, Cc, Cc, 0);
        attn_kernel<<<(Gg * Hh * (Ss / 16)) / 4, 128, 0, stream>>>(qb, kb, vb, ab);
        gemm(ab, WT[l][5], bo, h, h2, Nn, Cc, Cc, 0);            // + residual h
        bn(h2, h2, P(I_N2G) + l * Cc, P(I_N2B) + l * Cc, Nn, Cc, 0);

        // combine + FFN
        add_kernel<<<(Nn * Cc) / 256, 256, 0, stream>>>(h1, h2, outc, Nn * Cc);
        gemm(outc, WT[l][6], fb1, nullptr, f1, Nn, Cc, C2, 1);
        gemm(f1,   WT[l][7], fb2, outc,    h,  Nn, C2, Cc, 0);   // + residual outc
        bn(h, h, P(I_N3G) + l * Cc, P(I_N3B) + l * Cc, Nn, Cc, 0);
    }

    // ---------------- output: h + heads ----------------
    assemble_h_kernel<<<(Nn * Cc) / 256, 256, 0, stream>>>(h, out);
    const int head_od[3]  = {1, 1, 2};
    const int head_col[3] = {Cc, Cc + 1, Cc + 2};
    for (int hi = 0; hi < 3; ++hi) {
        int base = I_HEAD0 + hi * 10;   // w1,b1,g1,bn1,w2,b2,g2,bn2,w3,b3
        gemm(h,  WTh[hi][0], P(base + 1), nullptr, a1, Nn, Cc, 128, 0);
        bn(a1, a1, P(base + 2), P(base + 3), Nn, 128, 1);
        gemm(a1, WTh[hi][1], P(base + 5), nullptr, a2, Nn, 128, 128, 0);
        bn(a2, a2, P(base + 6), P(base + 7), Nn, 128, 1);
        head_out_kernel<<<(Nn + 255) / 256, 256, 0, stream>>>(
            a2, P(base + 8), P(base + 9), out, head_od[hi], head_col[hi]);
    }
}
